// GNN_Predictor_46694884442154
// MI455X (gfx1250) — matrix operand order repaired and checked by reading, not compiled, not run on hardware
//
#include <hip/hip_runtime.h>

typedef __attribute__((ext_vector_type(16))) _Float16 v16h;
typedef __attribute__((ext_vector_type(8)))  _Float16 v8h;
typedef __attribute__((ext_vector_type(8)))  float    v8f;

#define ST    68          // LDS row stride (floats), 64 + pad, multiple of 4
#define ROWS  128         // 32 graphs * 4 nodes per block
#define GPB   32          // graphs per block

// ---- A fragment (16x32 f16) from LDS f32 buffer, row-major stride ST ----
// caller passes kb = k0 + ((lane&16)?8:0); kb is a multiple of 8 -> float4 aligned.
__device__ inline v16h load_a_lds(const float* buf, int row, int kb) {
  const float4* p = (const float4*)(buf + row * ST + kb);
  float4 a0 = p[0], a1 = p[1];     // K = kb+0..7
  float4 a2 = p[4], a3 = p[5];     // K = kb+16..23
  v16h a;
  a[0]=(_Float16)a0.x; a[1]=(_Float16)a0.y; a[2]=(_Float16)a0.z; a[3]=(_Float16)a0.w;
  a[4]=(_Float16)a1.x; a[5]=(_Float16)a1.y; a[6]=(_Float16)a1.z; a[7]=(_Float16)a1.w;
  a[8]=(_Float16)a2.x; a[9]=(_Float16)a2.y; a[10]=(_Float16)a2.z; a[11]=(_Float16)a2.w;
  a[12]=(_Float16)a3.x; a[13]=(_Float16)a3.y; a[14]=(_Float16)a3.z; a[15]=(_Float16)a3.w;
  return a;
}

// ---- B fragment (32x16 f16) from LDS f16 weights W[n][k], row stride ldw halfs ----
// lane L: N = n0 + (L&15); elements e<8: K = kbase + (L<16?0:8) + e, e>=8: K = kbase + 16 + (L<16?0:8) + (e-8)
// ldw/kbase multiples of 8 halfs -> 16B aligned v8h loads (2x ds_load_b128).
__device__ inline v16h load_b_lds(const _Float16* Wl, int ldw, int n0, int kbase, int lane) {
  const _Float16* p = Wl + (n0 + (lane & 15)) * ldw + kbase + ((lane & 16) ? 8 : 0);
  v8h lo = *(const v8h*)p;
  v8h hi = *(const v8h*)(p + 16);
  return __builtin_shufflevector(lo, hi, 0,1,2,3,4,5,6,7,8,9,10,11,12,13,14,15);
}

// ---- stage contiguous f32 weights -> LDS f16 (totals multiple of 2048) ----
__device__ inline void stage_f16(const float* __restrict__ G, _Float16* L, int total, int tid) {
  for (int i = tid * 8; i < total; i += 256 * 8) {
    float4 a = *(const float4*)(G + i);
    float4 b = *(const float4*)(G + i + 4);
    v8h h;
    h[0]=(_Float16)a.x; h[1]=(_Float16)a.y; h[2]=(_Float16)a.z; h[3]=(_Float16)a.w;
    h[4]=(_Float16)b.x; h[5]=(_Float16)b.y; h[6]=(_Float16)b.z; h[7]=(_Float16)b.w;
    *(v8h*)(L + i) = h;
  }
}

// ---- stage kw[:,0:128] (row stride 133, unaligned rows) -> LDS f16 [64][128] ----
__device__ inline void stage_kwAB(const float* __restrict__ kw, _Float16* L, int tid) {
  for (int i = tid * 8; i < 64 * 128; i += 256 * 8) {
    int row = i >> 7, col = i & 127;
    const float* g = kw + row * 133 + col;
    v8h h;
#pragma unroll
    for (int e = 0; e < 8; ++e) h[e] = (_Float16)g[e];
    *(v8h*)(L + i) = h;
  }
}

// ---- [16 x 64] = A(rows m0..m0+15, K=64 f32 LDS) @ Wl^T (+bias_lds, opt lrelu) -> LDS O ----
__device__ inline void gemm64(const float* A, int m0,
                              const _Float16* Wl, int ldw, int kbase,
                              const float* bias_lds, int lrelu,
                              float* O, int lane) {
  int arow = m0 + (lane & 15);
  v16h a0 = load_a_lds(A, arow, 0  + ((lane & 16) ? 8 : 0));
  v16h a1 = load_a_lds(A, arow, 32 + ((lane & 16) ? 8 : 0));
#pragma unroll
  for (int t = 0; t < 4; ++t) {
    v8f c = {};
    v16h b0 = load_b_lds(Wl, ldw, t * 16, kbase, lane);
    v16h b1 = load_b_lds(Wl, ldw, t * 16, kbase + 32, lane);
    c = __builtin_amdgcn_wmma_f32_16x16x32_f16(false, a0, false, b0, (short)0, c, false, false);
    c = __builtin_amdgcn_wmma_f32_16x16x32_f16(false, a1, false, b1, (short)0, c, false, false);
    int n = t * 16 + (lane & 15);
    float bv = bias_lds[n];
    int rb = m0 + ((lane & 16) ? 8 : 0);
#pragma unroll
    for (int v = 0; v < 8; ++v) {
      float x = c[v] + bv;
      if (lrelu) x = x > 0.0f ? x : 0.01f * x;
      O[(rb + v) * ST + n] = x;
    }
  }
}

// same, but keep the 16x64 result in registers (4 C tiles)
__device__ inline void gemm64_regs(const float* A, int m0,
                                   const _Float16* Wl, int ldw, int kbase,
                                   const float* bias_lds, v8f* res, int lane) {
  int arow = m0 + (lane & 15);
  v16h a0 = load_a_lds(A, arow, 0  + ((lane & 16) ? 8 : 0));
  v16h a1 = load_a_lds(A, arow, 32 + ((lane & 16) ? 8 : 0));
#pragma unroll
  for (int t = 0; t < 4; ++t) {
    v8f c = {};
    v16h b0 = load_b_lds(Wl, ldw, t * 16, kbase, lane);
    v16h b1 = load_b_lds(Wl, ldw, t * 16, kbase + 32, lane);
    c = __builtin_amdgcn_wmma_f32_16x16x32_f16(false, a0, false, b0, (short)0, c, false, false);
    c = __builtin_amdgcn_wmma_f32_16x16x32_f16(false, a1, false, b1, (short)0, c, false, false);
    float bv = bias_lds[t * 16 + (lane & 15)];
#pragma unroll
    for (int v = 0; v < 8; ++v) res[t][v] = c[v] + bv;
  }
}

// ---- per-edge combine: H(=RES) += sum_{j!=nd} lrelu(P[dst] + Q[src] + ea@kwC) ----
__device__ inline void edge_combine(float* Hs, const float* Ps, const float* Qs,
                                    const float* eas, const float* kwcs, int tid) {
  for (int i = tid; i < ROWS * 64; i += 256) {
    int r = i >> 6, n = i & 63;
    int g = r >> 2, nd = r & 3;
    float p = Ps[r * ST + n];
    float acc = Hs[r * ST + n];
#pragma unroll
    for (int j = 0; j < 4; ++j) {
      if (j == nd) continue;
      int e = j * 3 + nd - (nd > j ? 1 : 0);     // pairs = [(i,j) i!=j], src=i, dst=j
      const float* ea = &eas[(g * 12 + e) * 5];
      float rt = 0.0f;
#pragma unroll
      for (int c = 0; c < 5; ++c) rt += ea[c] * kwcs[n * 5 + c];
      float m = p + Qs[(g * 4 + j) * ST + n] + rt;
      acc += m > 0.0f ? m : 0.01f * m;
    }
    Hs[r * ST + n] = acc;
  }
}

// ---- 32x64 MLP step (readout): 8 waves cover 2 M-tiles x 4 N-tiles ----
__device__ inline void mlp32x64(const float* A, float* O,
                                const _Float16* Wl, const float* bias_lds, int lrelu,
                                int wave, int lane) {
  int mt = wave >> 2, nt = wave & 3;
  int arow = mt * 16 + (lane & 15);
  v8f c = {};
  v16h a0 = load_a_lds(A, arow, 0  + ((lane & 16) ? 8 : 0));
  v16h a1 = load_a_lds(A, arow, 32 + ((lane & 16) ? 8 : 0));
  v16h b0 = load_b_lds(Wl, 64, nt * 16, 0, lane);
  v16h b1 = load_b_lds(Wl, 64, nt * 16, 32, lane);
  c = __builtin_amdgcn_wmma_f32_16x16x32_f16(false, a0, false, b0, (short)0, c, false, false);
  c = __builtin_amdgcn_wmma_f32_16x16x32_f16(false, a1, false, b1, (short)0, c, false, false);
  int n = nt * 16 + (lane & 15);
  float bv = bias_lds[n];
  int rb = mt * 16 + ((lane & 16) ? 8 : 0);
#pragma unroll
  for (int v = 0; v < 8; ++v) {
    float x = c[v] + bv;
    if (lrelu) x = x > 0.0f ? x : 0.01f * x;
    O[(rb + v) * ST + n] = x;
  }
}

__global__ __launch_bounds__(256) void gnn_fused_kernel(
    const float* __restrict__ xg, const float* __restrict__ eag,
    const float* __restrict__ c0_rw1, const float* __restrict__ c0_rb1,
    const float* __restrict__ c0_rw2, const float* __restrict__ c0_rb2,
    const float* __restrict__ c0_kw,
    const float* __restrict__ cl_rw1, const float* __restrict__ cl_rb1,
    const float* __restrict__ cl_rw2, const float* __restrict__ cl_rb2,
    const float* __restrict__ cl_kw,
    const float* __restrict__ fc3_w, const float* __restrict__ fc3_b,
    const float* __restrict__ m0_w, const float* __restrict__ m0_b,
    const float* __restrict__ mh_w, const float* __restrict__ mf_w,
    const float* __restrict__ mf_b,
    float* __restrict__ out)
{
  __shared__ float Hs[ROWS * ST];                        // features / RES / out (in place)
  __shared__ float Ps[ROWS * ST];                        // T1, then P, then mu/z
  __shared__ float Qs[ROWS * ST];                        // Q, then z ping-pong
  __shared__ __attribute__((aligned(32))) _Float16 Wh[64 * 256];  // staged f16 weights
  __shared__ float eas[384 * 5];                         // edge_attr for 32 graphs
  __shared__ float kwcs[64 * 5];                         // per-layer edge-col slice of kw
  __shared__ float xbs[ROWS * 4];                        // raw x for layer 0
  __shared__ float bias_a[64], bias_b[64], zbias[64], mfws[64];

  const int tid  = threadIdx.x;
  const int lane = tid & 31;
  const int wave = tid >> 5;
  const int gb   = blockIdx.x;
  const int rowbase = gb * ROWS;

  // ---------------- stage inputs ----------------
  for (int i = tid; i < ROWS * 4; i += 256) xbs[i] = xg[rowbase * 4 + i];
  for (int i = tid; i < 384 * 5; i += 256) eas[i] = eag[gb * 384 * 5 + i];
  for (int i = tid; i < 64 * 5;  i += 256) kwcs[i] = c0_kw[(i / 5) * 13 + 8 + (i % 5)];
  if (tid < 64) zbias[tid] = 0.0f;
  __syncthreads();

  // ---------------- layer 0 (input dim 4): VALU, K=4 below WMMA shape ----------
  for (int i = tid; i < ROWS * 64; i += 256) {
    int r = i >> 6, n = i & 63;
    const float* xr = &xbs[r * 4];
    float res = c0_rb2[n];
#pragma unroll
    for (int k = 0; k < 4; ++k) {
      float t1 = c0_rb1[k];
#pragma unroll
      for (int c = 0; c < 4; ++c) t1 += c0_rw1[k * 4 + c] * xr[c];
      res += c0_rw2[n * 4 + k] * t1;
    }
    float p = 0.0f, q = 0.0f;
#pragma unroll
    for (int c = 0; c < 4; ++c) {
      p += c0_kw[n * 13 + c] * xr[c];       // dst (x_i) columns 0..3
      q += c0_kw[n * 13 + 4 + c] * xr[c];   // src (x_j) columns 4..7
    }
    Hs[r * ST + n] = res;
    Ps[r * ST + n] = p;
    Qs[r * ST + n] = q;
  }
  __syncthreads();
  edge_combine(Hs, Ps, Qs, eas, kwcs, tid);
  __syncthreads();

  // ---------------- hidden conv layers 1..3 (WMMA) ----------------
  const int m0 = wave * 16;                 // each wave owns 16 rows = 4 graphs
  for (int l = 0; l < 3; ++l) {
    const float* rw1 = cl_rw1 + l * 4096;
    const float* rw2 = cl_rw2 + l * 4096;
    const float* kw  = cl_kw  + l * 64 * 133;

    // stage residual weights (f16), edge slice, biases
    stage_f16(rw1, Wh,        4096, tid);
    stage_f16(rw2, Wh + 4096, 4096, tid);
    for (int i = tid; i < 64 * 5; i += 256)
      kwcs[i] = kw[(i / 5) * 133 + 128 + (i % 5)];
    if (tid < 64) { bias_a[tid] = cl_rb1[l * 64 + tid]; bias_b[tid] = cl_rb2[l * 64 + tid]; }
    __syncthreads();

    // T1 = H @ rw1^T + rb1 -> Ps
    gemm64(Hs, m0, Wh, 64, 0, bias_a, 0, Ps, lane);
    __syncthreads();
    // RES = T1 @ rw2^T + rb2 -> registers
    v8f res[4];
    gemm64_regs(Ps, m0, Wh + 4096, 64, 0, bias_b, res, lane);
    __syncthreads();

    // stage kw[:,0:128] as f16 [64][128]
    stage_kwAB(kw, Wh, tid);
    __syncthreads();

    // P = H @ kw[:,0:64]^T -> Ps ;  Q = H @ kw[:,64:128]^T -> Qs
    gemm64(Hs, m0, Wh, 128, 0,  zbias, 0, Ps, lane);
    gemm64(Hs, m0, Wh, 128, 64, zbias, 0, Qs, lane);
    __syncthreads();

    // spill RES over H (H fully consumed)
    {
      int rb = m0 + ((lane & 16) ? 8 : 0);
#pragma unroll
      for (int t = 0; t < 4; ++t) {
        int n = t * 16 + (lane & 15);
#pragma unroll
        for (int v = 0; v < 8; ++v) Hs[(rb + v) * ST + n] = res[t][v];
      }
    }
    __syncthreads();
    edge_combine(Hs, Ps, Qs, eas, kwcs, tid);   // Hs := RES + agg
    __syncthreads();
  }

  // ---------------- readout: mu = g @ fc3_w^T + fc3_b  (M=32, K=256) ----------
  stage_f16(fc3_w, Wh, 64 * 256, tid);
  if (tid < 64) bias_a[tid] = fc3_b[tid];
  for (int i = tid; i < 64; i += 256) mfws[i] = mf_w[i];
  __syncthreads();
  {
    int mt = wave >> 2, nt = wave & 3;
    int bgl = mt * 16 + (lane & 15);       // graph index for this lane's A row
    v8f c = {};
#pragma unroll
    for (int ks = 0; ks < 8; ++ks) {
      int k0 = ks * 32;
      int node = k0 >> 6;                  // g[b][k] = H[4b + k/64][k%64]
      int kb = (k0 & 63) + ((lane & 16) ? 8 : 0);
      v16h a = load_a_lds(Hs, 4 * bgl + node, kb);
      v16h b = load_b_lds(Wh, 256, nt * 16, k0, lane);
      c = __builtin_amdgcn_wmma_f32_16x16x32_f16(false, a, false, b, (short)0, c, false, false);
    }
    int n = nt * 16 + (lane & 15);
    float bv = bias_a[n];
    int rb = mt * 16 + ((lane & 16) ? 8 : 0);
#pragma unroll
    for (int v = 0; v < 8; ++v) Ps[(rb + v) * ST + n] = c[v] + bv;   // mu rows 0..31
  }
  __syncthreads();

  // ---------------- readout MLP: lrelu chain, ping-pong Ps/Qs ----------------
  stage_f16(m0_w, Wh, 4096, tid);
  if (tid < 64) bias_a[tid] = m0_b[tid];
  __syncthreads();
  mlp32x64(Ps, Qs, Wh, bias_a, 1, wave, lane);
  __syncthreads();
  stage_f16(mh_w + 0 * 4096, Wh, 4096, tid);
  __syncthreads();
  mlp32x64(Qs, Ps, Wh, zbias, 1, wave, lane);
  __syncthreads();
  stage_f16(mh_w + 1 * 4096, Wh, 4096, tid);
  __syncthreads();
  mlp32x64(Ps, Qs, Wh, zbias, 1, wave, lane);
  __syncthreads();
  stage_f16(mh_w + 2 * 4096, Wh, 4096, tid);
  __syncthreads();
  mlp32x64(Qs, Ps, Wh, zbias, 1, wave, lane);
  __syncthreads();

  // ---------------- final head: [B,1] dot with mf_w ----------------
  if (tid < GPB) {
    float s = mf_b[0];
#pragma unroll 8
    for (int k = 0; k < 64; ++k) s += Ps[tid * ST + k] * mfws[k];
    out[gb * GPB + tid] = s;
  }
}

extern "C" void kernel_launch(void* const* d_in, const int* in_sizes, int n_in,
                              void* d_out, int out_size, void* d_ws, size_t ws_size,
                              hipStream_t stream) {
  (void)n_in; (void)d_ws; (void)ws_size; (void)in_sizes;
  const float* x       = (const float*)d_in[0];
  const float* ea      = (const float*)d_in[1];
  const float* c0_rw1  = (const float*)d_in[2];
  const float* c0_rb1  = (const float*)d_in[3];
  const float* c0_rw2  = (const float*)d_in[4];
  const float* c0_rb2  = (const float*)d_in[5];
  const float* c0_kw   = (const float*)d_in[6];
  const float* cl_rw1  = (const float*)d_in[7];
  const float* cl_rb1  = (const float*)d_in[8];
  const float* cl_rw2  = (const float*)d_in[9];
  const float* cl_rb2  = (const float*)d_in[10];
  const float* cl_kw   = (const float*)d_in[11];
  const float* fc3_w   = (const float*)d_in[12];
  const float* fc3_b   = (const float*)d_in[13];
  const float* m0_w    = (const float*)d_in[14];
  const float* m0_b    = (const float*)d_in[15];
  const float* mh_w    = (const float*)d_in[16];
  const float* mf_w    = (const float*)d_in[17];
  const float* mf_b    = (const float*)d_in[18];
  // d_in[19] = edge_index: topology is structurally known (complete 4-node graphs)

  int B = out_size;              // output is [B, 1] float32
  int blocks = B / GPB;          // 32 graphs per block
  gnn_fused_kernel<<<blocks, 256, 0, stream>>>(
      x, ea, c0_rw1, c0_rb1, c0_rw2, c0_rb2, c0_kw,
      cl_rw1, cl_rb1, cl_rw2, cl_rb2, cl_kw,
      fc3_w, fc3_b, m0_w, m0_b, mh_w, mf_w, mf_b,
      (float*)d_out);
}
